// CrissCrossAttention_25005299597573
// MI455X (gfx1250) — matrix operand order repaired and checked by reading, not compiled, not run on hardware
//
#include <hip/hip_runtime.h>
#include <hip/hip_bf16.h>

typedef __attribute__((ext_vector_type(16))) _Float16 v16h;
typedef __attribute__((ext_vector_type(8)))  _Float16 v8h;
typedef __attribute__((ext_vector_type(8)))  float    v8f;

constexpr int B_ = 8, H_ = 160, W_ = 160, C_ = 64, D_ = 8;
constexpr int NPOS = B_ * H_ * W_;            // 204800 positions

// Padded LDS leading dimensions (in halves) chosen for conflict-free access:
// byte strides 144 / 336 / 208 -> DWORD strides 36 / 84 / 52, all giving
// distinct banks across the 16 lanes of a half-wave (gcd arguments mod 64).
constexpr int LDX  = 72;    // x / weight tiles
constexpr int LDVT = 168;   // transposed V column/row
constexpr int LDP  = 104;   // probability slab

__device__ __forceinline__ v8f wmma_f16(v16h a, v16h b, v8f c) {
  return __builtin_amdgcn_wmma_f32_16x16x32_f16(false, a, false, b, (short)0, c, false, false);
}

__device__ __forceinline__ v16h concat8(v8h lo, v8h hi) {
  return __builtin_shufflevector(lo, hi, 0,1,2,3,4,5,6,7,8,9,10,11,12,13,14,15);
}

// A fragment (16x32 f16) from row-major LDS [rows][ld] at (row0, k0).
// lanes 0-15: M=lane, K=k0+{0..7,16..23}; lanes 16-31: M=lane-16, K=k0+{8..15,24..31}
__device__ __forceinline__ v16h ldsA(const _Float16* base, int ld, int row0, int k0, int lane) {
  const _Float16* p = base + (row0 + (lane & 15)) * ld + k0 + ((lane & 16) ? 8 : 0);
  const v8h lo = *(const v8h*)p;
  const v8h hi = *(const v8h*)(p + 16);
  return concat8(lo, hi);
}

// B fragment (32x16 f16) from COLUMN-major LDS [n][ld] at (k0, n0):
// lane's column is contiguous -> two ds_load_b128.
__device__ __forceinline__ v16h ldsBcol(const _Float16* base, int ld, int k0, int n0, int lane) {
  const _Float16* p = base + (n0 + (lane & 15)) * ld + k0 + ((lane & 16) ? 8 : 0);
  const v8h lo = *(const v8h*)p;
  const v8h hi = *(const v8h*)(p + 16);
  return concat8(lo, hi);
}

// ---------------------------------------------------------------------------
// Kernel 1: fused QKV projection. (204800 x 64) * (64 x [qk16 | v64]) -> f16
// q,k interleaved per position: qk[pos][0..7]=q, qk[pos][8..15]=k.
// ---------------------------------------------------------------------------
__global__ __launch_bounds__(256) void qkv_proj(
    const float* __restrict__ x,  const float* __restrict__ Wq,
    const float* __restrict__ Wk, const float* __restrict__ Wv,
    _Float16* __restrict__ qk, _Float16* __restrict__ vh)
{
  __shared__ alignas(16) _Float16 sX[128 * LDX];   // 18 KB
  __shared__ alignas(16) _Float16 sWT[80 * LDX];   // 11.5 KB, transposed [n][k]
  const int tid  = threadIdx.x;
  const int lane = tid & 31;
  const int wv   = tid >> 5;
  const int row0 = blockIdx.x * 128;

  // stage x: vectorized 8-float chunks -> 8 halves
  for (int i = tid; i < 128 * 8; i += 256) {
    const int row = i >> 3, c8 = (i & 7) * 8;
    const float* px = x + ((size_t)(row0 + row)) * 64 + c8;
    v8h hx;
#pragma unroll
    for (int j = 0; j < 8; ++j) hx[j] = (_Float16)px[j];
    *(v8h*)(sX + row * LDX + c8) = hx;
  }
  // stage weights transposed: sWT[n][k]
  for (int i = tid; i < 64 * 8; i += 256) {          // Wq
    const int kk = i >> 3, n = i & 7;
    sWT[n * LDX + kk] = (_Float16)Wq[kk * 8 + n];
  }
  for (int i = tid; i < 64 * 8; i += 256) {          // Wk
    const int kk = i >> 3, n = i & 7;
    sWT[(8 + n) * LDX + kk] = (_Float16)Wk[kk * 8 + n];
  }
  for (int i = tid; i < 64 * 64; i += 256) {         // Wv (contiguous reads)
    const int kk = i >> 6, c = i & 63;
    sWT[(16 + c) * LDX + kk] = (_Float16)Wv[i];
  }
  __syncthreads();

#pragma unroll
  for (int nt = 0; nt < 5; ++nt) {                   // N = 80 -> 5 tiles
    v8f acc = {};
#pragma unroll
    for (int kt = 0; kt < 2; ++kt) {                 // K = 64 -> 2 steps
      v16h a = ldsA(sX, LDX, wv * 16, kt * 32, lane);
      v16h b = ldsBcol(sWT, LDX, kt * 32, nt * 16, lane);
      acc = wmma_f16(a, b, acc);
    }
    const int n = nt * 16 + (lane & 15);
#pragma unroll
    for (int r = 0; r < 8; ++r) {
      const size_t m = (size_t)row0 + wv * 16 + ((lane & 16) ? 8 + r : r);
      const _Float16 val = (_Float16)acc[r];
      if (nt == 0) qk[m * 16 + n] = val;             // compile-time branch
      else         vh[m * 64 + (n - 16)] = val;
    }
  }
}

// ---------------------------------------------------------------------------
// Kernel 2: column pass. One workgroup per (b, w); 10 waves x 16 rows = 160.
// E_H = Q_col (160x8) * K_col^T (diag-masked); store m_H, s_H and
// unnormalized U_H = exp(E_H - m_H) * V_col (f32).
// ---------------------------------------------------------------------------
__global__ __launch_bounds__(320) void col_pass(
    const _Float16* __restrict__ qk, const _Float16* __restrict__ vh,
    float* __restrict__ UH, float* __restrict__ mH, float* __restrict__ sH)
{
  __shared__ alignas(16) _Float16 sK[H_ * D_];        // 2.5 KB [h][d]
  __shared__ alignas(16) _Float16 sVT[C_ * LDVT];     // 21 KB  [c][kh] transposed
  __shared__ alignas(16) _Float16 sP[10 * 16 * LDP];  // 32.5 KB per-wave prob slabs

  const int tid = threadIdx.x, lane = tid & 31, wv = tid >> 5;
  const int b = blockIdx.x / W_, w = blockIdx.x % W_;
  const bool up = (lane & 16) != 0;
  const int  m0 = wv * 16;
  const int  mbase = m0 + (up ? 8 : 0);
  const size_t colbase = (size_t)(b * H_) * W_ + w;

  if (tid == 0) {
    __builtin_prefetch(vh + colbase * 64, 0, 0);
    __builtin_prefetch(qk + colbase * 16, 0, 0);
  }
  // stage k column: one b128 per row
  for (int c = tid; c < H_; c += 320)
    *(v8h*)(sK + c * 8) = *(const v8h*)(qk + (colbase + (size_t)c * W_) * 16 + 8);
  // stage v column transposed
  for (int i = tid; i < H_ * 8; i += 320) {
    const int hh = i >> 3, c8 = (i & 7) * 8;
    const v8h vv = *(const v8h*)(vh + (colbase + (size_t)hh * W_) * 64 + c8);
#pragma unroll
    for (int j = 0; j < 8; ++j) sVT[(c8 + j) * LDVT + hh] = vv[j];
  }
  __syncthreads();

  // A fragment: Q rows, K=8 zero-padded to 32.
  const v8h zz = {};
  const int mq = m0 + (lane & 15);
  v8h qv = *(const v8h*)(qk + (colbase + (size_t)mq * W_) * 16);
  if (up) qv = zz;
  const v16h afrag = concat8(qv, zz);

  v8f acc[10];                                        // E_H slab 16x160
#pragma unroll
  for (int nt = 0; nt < 10; ++nt) {
    const int n = nt * 16 + (lane & 15);
    v8h kv = *(const v8h*)(sK + n * 8);
    if (up) kv = zz;
    const v16h bfrag = concat8(kv, zz);
    v8f z = {};
    acc[nt] = wmma_f16(afrag, bfrag, z);
  }

  // Diagonal mask + per-row max / sum(exp).
  float mrow[8];
#pragma unroll
  for (int r = 0; r < 8; ++r) {
    const int m = mbase + r;
    float mx = -__builtin_inff();
#pragma unroll
    for (int nt = 0; nt < 10; ++nt) {
      if (nt * 16 + (lane & 15) == m) acc[nt][r] = -__builtin_inff();
      mx = fmaxf(mx, acc[nt][r]);
    }
#pragma unroll
    for (int off = 1; off < 16; off <<= 1) mx = fmaxf(mx, __shfl_xor(mx, off, 32));
    float s = 0.f;
#pragma unroll
    for (int nt = 0; nt < 10; ++nt) s += __expf(acc[nt][r] - mx);
#pragma unroll
    for (int off = 1; off < 16; off <<= 1) s += __shfl_xor(s, off, 32);
    mrow[r] = mx;
    if ((lane & 15) == 0) {
      const size_t p = (size_t)(b * H_ + m) * W_ + w;
      mH[p] = mx; sH[p] = s;
    }
  }

  // U_H = P(160x160) * V_col(160x64), P via per-wave LDS slab (K split 96+64).
  _Float16* slab = sP + wv * (16 * LDP);
  v8f accu[4];
  { v8f z = {}; for (int i = 0; i < 4; ++i) accu[i] = z; }
  const int segTiles[2] = {6, 4};
  const int segKst[2]   = {3, 2};
#pragma unroll
  for (int seg = 0; seg < 2; ++seg) {
    for (int ntL = 0; ntL < segTiles[seg]; ++ntL) {
      const int gt  = seg * 6 + ntL;
      const int col = ntL * 16 + (lane & 15);
#pragma unroll
      for (int r = 0; r < 8; ++r)
        slab[((up ? 8 : 0) + r) * LDP + col] = (_Float16)__expf(acc[gt][r] - mrow[r]);
    }
    for (int kt = 0; kt < segKst[seg]; ++kt) {
      v16h a = ldsA(slab, LDP, 0, kt * 32, lane);
#pragma unroll
      for (int nt = 0; nt < 4; ++nt) {
        v16h bb = ldsBcol(sVT, LDVT, seg * 96 + kt * 32, nt * 16, lane);
        accu[nt] = wmma_f16(a, bb, accu[nt]);
      }
    }
  }

#pragma unroll
  for (int nt = 0; nt < 4; ++nt) {
    const int c = nt * 16 + (lane & 15);
#pragma unroll
    for (int r = 0; r < 8; ++r) {
      const int m = mbase + r;
      UH[((size_t)(b * H_ + m) * W_ + w) * C_ + c] = accu[nt][r];
    }
  }
}

// ---------------------------------------------------------------------------
// Kernel 3: row pass + softmax merge + epilogue. One workgroup per (b, h).
// ---------------------------------------------------------------------------
__global__ __launch_bounds__(320) void row_pass(
    const float* __restrict__ x,
    const _Float16* __restrict__ qk, const _Float16* __restrict__ vh,
    const float* __restrict__ UH, const float* __restrict__ mH,
    const float* __restrict__ sH,
    const float* __restrict__ gamma, float* __restrict__ out)
{
  __shared__ alignas(16) _Float16 sK[W_ * D_];
  __shared__ alignas(16) _Float16 sVT[C_ * LDVT];
  __shared__ alignas(16) _Float16 sP[10 * 16 * LDP];

  const int tid = threadIdx.x, lane = tid & 31, wv = tid >> 5;
  const int b = blockIdx.x / H_, h = blockIdx.x % H_;
  const bool up = (lane & 16) != 0;
  const int  m0 = wv * 16;                   // M indexes w-positions
  const int  mbase = m0 + (up ? 8 : 0);
  const size_t rowbase = (size_t)(b * H_ + h) * W_;

  if (tid == 0) {
    __builtin_prefetch(vh + rowbase * 64, 0, 0);
    __builtin_prefetch(UH + rowbase * 64, 0, 0);
  }
  for (int c = tid; c < W_; c += 320)
    *(v8h*)(sK + c * 8) = *(const v8h*)(qk + (rowbase + c) * 16 + 8);
  for (int i = tid; i < W_ * 8; i += 320) {
    const int wp = i >> 3, c8 = (i & 7) * 8;
    const v8h vv = *(const v8h*)(vh + (rowbase + wp) * 64 + c8);
#pragma unroll
    for (int j = 0; j < 8; ++j) sVT[(c8 + j) * LDVT + wp] = vv[j];
  }
  __syncthreads();

  const v8h zz = {};
  const int mq = m0 + (lane & 15);
  v8h qv = *(const v8h*)(qk + (rowbase + mq) * 16);
  if (up) qv = zz;
  const v16h afrag = concat8(qv, zz);

  v8f acc[10];                               // E_W slab (no diagonal mask)
#pragma unroll
  for (int nt = 0; nt < 10; ++nt) {
    const int n = nt * 16 + (lane & 15);
    v8h kv = *(const v8h*)(sK + n * 8);
    if (up) kv = zz;
    const v16h bfrag = concat8(kv, zz);
    v8f z = {};
    acc[nt] = wmma_f16(afrag, bfrag, z);
  }

  float mrow[8], srow[8];
#pragma unroll
  for (int r = 0; r < 8; ++r) {
    float mx = -__builtin_inff();
#pragma unroll
    for (int nt = 0; nt < 10; ++nt) mx = fmaxf(mx, acc[nt][r]);
#pragma unroll
    for (int off = 1; off < 16; off <<= 1) mx = fmaxf(mx, __shfl_xor(mx, off, 32));
    float s = 0.f;
#pragma unroll
    for (int nt = 0; nt < 10; ++nt) s += __expf(acc[nt][r] - mx);
#pragma unroll
    for (int off = 1; off < 16; off <<= 1) s += __shfl_xor(s, off, 32);
    mrow[r] = mx; srow[r] = s;
  }

  _Float16* slab = sP + wv * (16 * LDP);
  v8f accu[4];
  { v8f z = {}; for (int i = 0; i < 4; ++i) accu[i] = z; }
  const int segTiles[2] = {6, 4};
  const int segKst[2]   = {3, 2};
#pragma unroll
  for (int seg = 0; seg < 2; ++seg) {
    for (int ntL = 0; ntL < segTiles[seg]; ++ntL) {
      const int gt  = seg * 6 + ntL;
      const int col = ntL * 16 + (lane & 15);
#pragma unroll
      for (int r = 0; r < 8; ++r)
        slab[((up ? 8 : 0) + r) * LDP + col] = (_Float16)__expf(acc[gt][r] - mrow[r]);
    }
    for (int kt = 0; kt < segKst[seg]; ++kt) {
      v16h a = ldsA(slab, LDP, 0, kt * 32, lane);
#pragma unroll
      for (int nt = 0; nt < 4; ++nt) {
        v16h bb = ldsBcol(sVT, LDVT, seg * 96 + kt * 32, nt * 16, lane);
        accu[nt] = wmma_f16(a, bb, accu[nt]);
      }
    }
  }

  // Merge the two softmax halves; write final output.
  const float g = gamma[0];
#pragma unroll
  for (int r = 0; r < 8; ++r) {
    const int wp = mbase + r;
    const size_t p = rowbase + wp;
    const float mh = mH[p], sh = sH[p];
    const float M  = fmaxf(mh, mrow[r]);
    const float ea = __expf(mh - M);
    const float eb = __expf(mrow[r] - M);
    const float inv = 1.f / (sh * ea + srow[r] * eb);
#pragma unroll
    for (int nt = 0; nt < 4; ++nt) {
      const int c = nt * 16 + (lane & 15);
      const size_t idx = p * C_ + c;
      out[idx] = g * (UH[idx] * ea + accu[nt][r] * eb) * inv + x[idx];
    }
  }
}

// ---------------------------------------------------------------------------
extern "C" void kernel_launch(void* const* d_in, const int* in_sizes, int n_in,
                              void* d_out, int out_size, void* d_ws, size_t ws_size,
                              hipStream_t stream) {
  (void)in_sizes; (void)n_in; (void)out_size; (void)ws_size;
  const float* x     = (const float*)d_in[0];
  const float* Wq    = (const float*)d_in[1];
  const float* Wk    = (const float*)d_in[2];
  const float* Wv    = (const float*)d_in[3];
  const float* gamma = (const float*)d_in[4];
  float* out = (float*)d_out;

  char* ws = (char*)d_ws;
  constexpr size_t OFF_QK = 0;                                 // [NPOS][16] f16
  constexpr size_t OFF_V  = OFF_QK + (size_t)NPOS * 16 * 2;    //  6,553,600
  constexpr size_t OFF_UH = OFF_V  + (size_t)NPOS * C_ * 2;    // 32,768,000
  constexpr size_t OFF_MH = OFF_UH + (size_t)NPOS * C_ * 4;    // 85,196,800
  constexpr size_t OFF_SH = OFF_MH + (size_t)NPOS * 4;         // 86,016,000
  _Float16* qk = (_Float16*)(ws + OFF_QK);
  _Float16* vh = (_Float16*)(ws + OFF_V);
  float*    UH = (float*)(ws + OFF_UH);
  float*    mH = (float*)(ws + OFF_MH);
  float*    sH = (float*)(ws + OFF_SH);

  qkv_proj<<<NPOS / 128, 256, 0, stream>>>(x, Wq, Wk, Wv, qk, vh);
  col_pass<<<B_ * W_, 320, 0, stream>>>(qk, vh, UH, mH, sH);
  row_pass<<<B_ * H_, 320, 0, stream>>>(x, qk, vh, UH, mH, sH, gamma, out);
}
